// HybridGNN_80487687127283
// MI455X (gfx1250) — matrix operand order repaired and checked by reading, not compiled, not run on hardware
//
#include <hip/hip_runtime.h>
#include <hip/hip_bf16.h>
#include <stdint.h>

#define IN_DIM 128
#define HID    256
#define OUTD   256
#define FUSK   768   // 3*HID

typedef __attribute__((ext_vector_type(16))) __bf16   v16bf;
typedef __attribute__((ext_vector_type(8)))  float    v8f;
typedef __attribute__((ext_vector_type(4)))  uint32_t u32x4;

// ---------- helpers ----------
__device__ __forceinline__ uint16_t f2bf(float f) {
  uint32_t u = __float_as_uint(f);
  u += 0x7FFFu + ((u >> 16) & 1u);          // round-to-nearest-even
  return (uint16_t)(u >> 16);
}
__device__ __forceinline__ float lrelu02(float x) { return x > 0.f ? x : 0.2f * x; }
// order-preserving float <-> uint encoding for atomicMax over signed floats
__device__ __forceinline__ unsigned fenc(float f) {
  unsigned u = __float_as_uint(f);
  return (u & 0x80000000u) ? ~u : (u | 0x80000000u);
}
__device__ __forceinline__ float fdec(unsigned s) {
  unsigned u = (s & 0x80000000u) ? (s & 0x7FFFFFFFu) : ~s;
  return __uint_as_float(u);
}

// ---------- generic bf16 WMMA GEMM ----------
// C[M x Nc] (f32) = A[M x K](bf16 row-major) * B (bf16 packed as [Nc][K], each output
// column's K values contiguous) [+ C if accumulate] [+ bias[col]]
// One wave computes one 16x16 tile; 8 waves per 256-thread block. Requires M%16==0,
// K%32==0, Nc%16==0 (true for all shapes here: M=50000, K=128/768, Nc=256).
// All operands stream through the 192MB L2; direct global fragment loads are the
// optimal path on MI455X for these panel sizes (no LDS staging needed).
__global__ void __launch_bounds__(256)
gemm_bf16_wmma(const uint16_t* __restrict__ A, const uint16_t* __restrict__ Bp,
               float* __restrict__ C, const float* __restrict__ bias,
               int M, int K, int Nc, int accumulate)
{
  const int wave  = threadIdx.x >> 5;
  const int lane  = threadIdx.x & 31;
  const int tilesN = Nc >> 4;
  const int tile   = blockIdx.x * (blockDim.x >> 5) + wave;
  const int tM = tile / tilesN;
  const int tN = tile - tM * tilesN;
  if (tM * 16 >= M) return;                 // uniform per wave -> EXEC stays all-ones

  const int half = lane >> 4;               // 0: lanes 0-15, 1: lanes 16-31
  const int mr   = lane & 15;
  const uint16_t* arow = A  + (size_t)(tM * 16 + mr) * K;   // A row for this lane
  const uint16_t* bcol = Bp + (size_t)(tN * 16 + mr) * K;   // B column for this lane

  v8f c = {};
  if (accumulate) {
#pragma unroll
    for (int r = 0; r < 8; ++r) {
      int row = tM * 16 + r + half * 8;
      int col = tN * 16 + mr;
      c[r] = C[(size_t)row * Nc + col];
    }
  }

  for (int k0 = 0; k0 < K; k0 += 32) {
    union { v16bf v; u32x4 q[2]; } a, b;
    // A 16x32 bf16 layout (ISA 7.12.2): elems0-7 = K[k0 + 8*half .. +7],
    //                                   elems8-15 = K[k0+16 + 8*half .. +7]
    a.q[0] = *(const u32x4*)(arow + k0 + half * 8);
    a.q[1] = *(const u32x4*)(arow + k0 + 16 + half * 8);
    // B 32x16 bf16: lane holds one column, same K order as A (k0 + 8*half .., k0+16 + 8*half ..)
    b.q[0] = *(const u32x4*)(bcol + k0 + half * 8);
    b.q[1] = *(const u32x4*)(bcol + k0 + 16 + half * 8);
    c = __builtin_amdgcn_wmma_f32_16x16x32_bf16(false, a.v, false, b.v,
                                                (short)0, c, false, false);
  }

#pragma unroll
  for (int r = 0; r < 8; ++r) {
    int row = tM * 16 + r + half * 8;       // C/D layout: M = r + 8*half, N = mr
    int col = tN * 16 + mr;
    float v = c[r];
    if (bias) v += bias[col];
    C[(size_t)row * Nc + col] = v;
  }
}

// ---------- packing / casting ----------
__global__ void cast_f32_bf16(const float* __restrict__ src, uint16_t* __restrict__ dst, int n) {
  int i = blockIdx.x * blockDim.x + threadIdx.x;
  if (i < n) dst[i] = f2bf(src[i]);
}
// W [K x Nc] row-major f32  ->  [Nc x K] bf16 (column of B contiguous in K)
__global__ void pack_wT(const float* __restrict__ W, uint16_t* __restrict__ out, int K, int Nc) {
  int idx = blockIdx.x * blockDim.x + threadIdx.x;
  if (idx >= K * Nc) return;
  int k = idx / Nc, n = idx - k * Nc;
  out[(size_t)n * K + k] = f2bf(W[idx]);
}

// ---------- graph preprocessing ----------
__global__ void init_buffers(float* __restrict__ deg, float* __restrict__ hgcn,
                             float* __restrict__ msum, int N) {
  int i = blockIdx.x * blockDim.x + threadIdx.x;
  if (i < N * HID)    hgcn[i] = 0.f;
  if (i < N * IN_DIM) msum[i] = 0.f;
  if (i < N)          deg[i]  = 1.0f;      // self-loop included
}
__global__ void count_edges(const int* __restrict__ ei, float* __restrict__ deg, int E) {
  int e = blockIdx.x * blockDim.x + threadIdx.x;
  if (e < E) atomicAdd(&deg[ei[E + e]], 1.0f);   // exact: counts << 2^24
}

// ---------- GAT scalar stage ----------
__global__ void __launch_bounds__(256)
att_dots(const float* __restrict__ xa, const float* __restrict__ asv,
         const float* __restrict__ adv, float* __restrict__ asrc,
         float* __restrict__ adst, int N) {
  int gwave = (blockIdx.x * blockDim.x + threadIdx.x) >> 5;
  int lane  = threadIdx.x & 31;
  if (gwave >= N) return;
  const float* rowp = xa + (size_t)gwave * HID;
  float s = 0.f, d = 0.f;
  for (int t = lane; t < HID; t += 32) { float v = rowp[t]; s += v * asv[t]; d += v * adv[t]; }
  for (int off = 16; off > 0; off >>= 1) {
    s += __shfl_down(s, off, 32);
    d += __shfl_down(d, off, 32);
  }
  if (lane == 0) { asrc[gwave] = s; adst[gwave] = d; }
}
__global__ void gat_init(const float* __restrict__ asrc, const float* __restrict__ adst,
                         unsigned* __restrict__ menc, int N) {
  int i = blockIdx.x * blockDim.x + threadIdx.x;
  if (i < N) menc[i] = fenc(lrelu02(asrc[i] + adst[i]));    // self-loop seeds the max
}
__global__ void gat_max(const int* __restrict__ ei, const float* __restrict__ asrc,
                        const float* __restrict__ adst, unsigned* __restrict__ menc, int E) {
  int e = blockIdx.x * blockDim.x + threadIdx.x;
  if (e >= E) return;
  int row = ei[e], col = ei[E + e];
  atomicMax(&menc[col], fenc(lrelu02(asrc[row] + adst[col])));
}
// per-node self-loop softmax term: seeds hgat accumulator and z
__global__ void gat_mid(const float* __restrict__ asrc, const float* __restrict__ adst,
                        const unsigned* __restrict__ menc, const float* __restrict__ xa,
                        float* __restrict__ hgat, float* __restrict__ z, int N) {
  int idx = blockIdx.x * blockDim.x + threadIdx.x;
  if (idx >= N * HID) return;
  int i = idx / HID, t = idx - i * HID;
  float p = expf(lrelu02(asrc[i] + adst[i]) - fdec(menc[i]));  // self-loop term
  hgat[idx] = p * xa[idx];
  if (t == 0) z[i] = p;
}

// ---------- fused GCN+GAT edge propagation ----------
// One block per edge, 64 threads x float4 = 256 dims. Reads edge scalars once,
// streams both xg[row] and xa[row] rows (b128 loads), scalar f32 atomics into
// both accumulators. This pass is the HBM/L2-atomic bandwidth floor (~4GB).
__global__ void __launch_bounds__(64)
edge_prop(const int* __restrict__ ei, const float* __restrict__ xg,
          const float* __restrict__ xa, const float* __restrict__ deg,
          const unsigned* __restrict__ menc, const float* __restrict__ asrc,
          const float* __restrict__ adst, float* __restrict__ hgcn,
          float* __restrict__ hgat, float* __restrict__ z, int E) {
  int e = blockIdx.x;
  if (e >= E) return;
  int row = ei[e], col = ei[E + e];
  float norm = rsqrtf(deg[row]) * rsqrtf(deg[col]);
  float p    = expf(lrelu02(asrc[row] + adst[col]) - fdec(menc[col]));
  int t = threadIdx.x;
  const float4 g = *(const float4*)(xg + (size_t)row * HID + t * 4);
  const float4 a = *(const float4*)(xa + (size_t)row * HID + t * 4);
  float* pg = hgcn + (size_t)col * HID + t * 4;
  float* pa = hgat + (size_t)col * HID + t * 4;
  atomicAdd(pg + 0, norm * g.x); atomicAdd(pg + 1, norm * g.y);
  atomicAdd(pg + 2, norm * g.z); atomicAdd(pg + 3, norm * g.w);
  atomicAdd(pa + 0, p * a.x);    atomicAdd(pa + 1, p * a.y);
  atomicAdd(pa + 2, p * a.z);    atomicAdd(pa + 3, p * a.w);
  if (t == 0) atomicAdd(&z[col], p);
}

// ---------- SAGE ----------
__global__ void __launch_bounds__(32)
sage_edge(const int* __restrict__ ei, const float* __restrict__ x,
          float* __restrict__ msum, int E) {
  int e = blockIdx.x;
  if (e >= E) return;
  int row = ei[e], col = ei[E + e];
  int t = threadIdx.x;                       // 32 threads x float4 = 128 dims
  const float4 v = *(const float4*)(x + (size_t)row * IN_DIM + t * 4);
  float* pm = msum + (size_t)col * IN_DIM + t * 4;
  atomicAdd(pm + 0, v.x); atomicAdd(pm + 1, v.y);
  atomicAdd(pm + 2, v.z); atomicAdd(pm + 3, v.w);
}
__global__ void sage_mean(const float* __restrict__ msum, const float* __restrict__ deg,
                          uint16_t* __restrict__ meanbf, int N) {
  int idx = blockIdx.x * blockDim.x + threadIdx.x;
  if (idx >= N * IN_DIM) return;
  int i = idx >> 7;
  float cnt = fmaxf(deg[i] - 1.0f, 1.0f);   // deg counted with self-loop; SAGE has none
  meanbf[idx] = f2bf(msum[idx] / cnt);
}

// ---------- finalize: relu + bf16 write into fused feature matrix ----------
__global__ void fin_gcn(const float* __restrict__ hgcn, const float* __restrict__ xg,
                        const float* __restrict__ deg, const float* __restrict__ bgcn,
                        uint16_t* __restrict__ hbf, int N) {
  int idx = blockIdx.x * blockDim.x + threadIdx.x;
  if (idx >= N * HID) return;
  int i = idx / HID, t = idx - i * HID;
  float v = hgcn[idx] + xg[idx] / deg[i] + bgcn[t];   // self-loop norm = 1/deg
  hbf[(size_t)i * FUSK + t] = f2bf(fmaxf(v, 0.f));
}
__global__ void fin_gat(const float* __restrict__ hgat, const float* __restrict__ z,
                        const float* __restrict__ bgat, uint16_t* __restrict__ hbf, int N) {
  int idx = blockIdx.x * blockDim.x + threadIdx.x;
  if (idx >= N * HID) return;
  int i = idx / HID, t = idx - i * HID;
  float v = hgat[idx] / (z[i] + 1e-16f) + bgat[t];
  hbf[(size_t)i * FUSK + HID + t] = f2bf(fmaxf(v, 0.f));
}
__global__ void fin_sage(const float* __restrict__ hsage, const float* __restrict__ bsl,
                         uint16_t* __restrict__ hbf, int N) {
  int idx = blockIdx.x * blockDim.x + threadIdx.x;
  if (idx >= N * HID) return;
  int i = idx / HID, t = idx - i * HID;
  float v = hsage[idx] + bsl[t];
  hbf[(size_t)i * FUSK + 2 * HID + t] = f2bf(fmaxf(v, 0.f));
}

// ---------- launch ----------
extern "C" void kernel_launch(void* const* d_in, const int* in_sizes, int n_in,
                              void* d_out, int out_size, void* d_ws, size_t ws_size,
                              hipStream_t stream) {
  (void)n_in; (void)out_size; (void)ws_size;
  const float* x       = (const float*)d_in[0];
  const int*   ei      = (const int*)  d_in[1];
  const float* W_gcn   = (const float*)d_in[2];
  const float* b_gcn   = (const float*)d_in[3];
  const float* W_gat   = (const float*)d_in[4];
  const float* att_src = (const float*)d_in[5];
  const float* att_dst = (const float*)d_in[6];
  const float* b_gat   = (const float*)d_in[7];
  const float* W_sl    = (const float*)d_in[8];
  const float* b_sl    = (const float*)d_in[9];
  const float* W_sr    = (const float*)d_in[10];
  const float* W_fus   = (const float*)d_in[11];
  const float* b_fus   = (const float*)d_in[12];

  const int N = in_sizes[0] / IN_DIM;       // 50000 (multiple of 16)
  const int E = in_sizes[1] / 2;            // 800000

  // workspace carve-out (256B aligned slices)
  char* base = (char*)d_ws;
  size_t off = 0;
  auto alloc = [&](size_t bytes) -> void* {
    void* p = base + off;
    off = (off + bytes + 255) & ~(size_t)255;
    return p;
  };
  float*    xg      = (float*)   alloc((size_t)N * HID * 4);
  float*    xa      = (float*)   alloc((size_t)N * HID * 4);
  float*    hsage   = (float*)   alloc((size_t)N * HID * 4);
  float*    hgcn    = (float*)   alloc((size_t)N * HID * 4);
  float*    hgat    = (float*)   alloc((size_t)N * HID * 4);
  float*    msum    = (float*)   alloc((size_t)N * IN_DIM * 4);
  float*    deg     = (float*)   alloc((size_t)N * 4);
  float*    asrc    = (float*)   alloc((size_t)N * 4);
  float*    adst    = (float*)   alloc((size_t)N * 4);
  unsigned* menc    = (unsigned*)alloc((size_t)N * 4);
  float*    z       = (float*)   alloc((size_t)N * 4);
  uint16_t* xbf     = (uint16_t*)alloc((size_t)N * IN_DIM * 2);
  uint16_t* meanbf  = (uint16_t*)alloc((size_t)N * IN_DIM * 2);
  uint16_t* hbf     = (uint16_t*)alloc((size_t)N * FUSK * 2);
  uint16_t* wgcn_p  = (uint16_t*)alloc((size_t)IN_DIM * HID * 2);
  uint16_t* wgat_p  = (uint16_t*)alloc((size_t)IN_DIM * HID * 2);
  uint16_t* wsl_p   = (uint16_t*)alloc((size_t)IN_DIM * HID * 2);
  uint16_t* wsr_p   = (uint16_t*)alloc((size_t)IN_DIM * HID * 2);
  uint16_t* wfus_p  = (uint16_t*)alloc((size_t)FUSK * OUTD * 2);

  const int TB = 256;
  auto grid1 = [&](long n) { return (int)((n + TB - 1) / TB); };

  // init + degrees (deg includes self-loop)
  init_buffers<<<grid1((long)N * HID), TB, 0, stream>>>(deg, hgcn, msum, N);
  count_edges <<<grid1(E), TB, 0, stream>>>(ei, deg, E);

  // pack weights / cast inputs to bf16
  cast_f32_bf16<<<grid1((long)N * IN_DIM), TB, 0, stream>>>(x, xbf, N * IN_DIM);
  pack_wT<<<grid1(IN_DIM * HID), TB, 0, stream>>>(W_gcn, wgcn_p, IN_DIM, HID);
  pack_wT<<<grid1(IN_DIM * HID), TB, 0, stream>>>(W_gat, wgat_p, IN_DIM, HID);
  pack_wT<<<grid1(IN_DIM * HID), TB, 0, stream>>>(W_sl,  wsl_p,  IN_DIM, HID);
  pack_wT<<<grid1(IN_DIM * HID), TB, 0, stream>>>(W_sr,  wsr_p,  IN_DIM, HID);
  pack_wT<<<grid1(FUSK * OUTD),  TB, 0, stream>>>(W_fus, wfus_p, FUSK,   OUTD);

  // dense transforms on the matrix pipe
  const int tiles_h = (N / 16) * (HID / 16);
  const int gblocks = (tiles_h + 7) / 8;    // 8 waves per block
  gemm_bf16_wmma<<<gblocks, 256, 0, stream>>>(xbf, wgcn_p, xg,    nullptr, N, IN_DIM, HID, 0);
  gemm_bf16_wmma<<<gblocks, 256, 0, stream>>>(xbf, wgat_p, xa,    nullptr, N, IN_DIM, HID, 0);
  gemm_bf16_wmma<<<gblocks, 256, 0, stream>>>(xbf, wsr_p,  hsage, nullptr, N, IN_DIM, HID, 0);

  // GAT scalar stage: attention dots, stable segment-max, self-loop seeding
  att_dots<<<grid1((long)N * 32), TB, 0, stream>>>(xa, att_src, att_dst, asrc, adst, N);
  gat_init<<<grid1(N), TB, 0, stream>>>(asrc, adst, menc, N);
  gat_max <<<grid1(E), TB, 0, stream>>>(ei, asrc, adst, menc, E);
  gat_mid <<<grid1((long)N * HID), TB, 0, stream>>>(asrc, adst, menc, xa, hgat, z, N);

  // fused GCN+GAT feature propagation (one sweep over edges)
  edge_prop<<<E, 64, 0, stream>>>(ei, xg, xa, deg, menc, asrc, adst, hgcn, hgat, z, E);

  // SAGE: mean aggregate then second GEMM accumulating into hsage
  sage_edge<<<E, 32, 0, stream>>>(ei, x, msum, E);
  sage_mean<<<grid1((long)N * IN_DIM), TB, 0, stream>>>(msum, deg, meanbf, N);
  gemm_bf16_wmma<<<gblocks, 256, 0, stream>>>(meanbf, wsl_p, hsage, nullptr, N, IN_DIM, HID, 1);

  // finalize branches: relu + write bf16 into fused feature matrix [N x 768]
  fin_gcn <<<grid1((long)N * HID), TB, 0, stream>>>(hgcn, xg, deg, b_gcn, hbf, N);
  fin_gat <<<grid1((long)N * HID), TB, 0, stream>>>(hgat, z, b_gat, hbf, N);
  fin_sage<<<grid1((long)N * HID), TB, 0, stream>>>(hsage, b_sl, hbf, N);

  // fusion GEMM: out = hbf[N x 768] @ W_fus + b_fus  (f32 output)
  const int tiles_o = (N / 16) * (OUTD / 16);
  gemm_bf16_wmma<<<(tiles_o + 7) / 8, 256, 0, stream>>>(hbf, wfus_p, (float*)d_out,
                                                        b_fus, N, FUSK, OUTD, 0);
}
